// MiniMaxSparseMoeBlock_43963285242496
// MI455X (gfx1250) — matrix operand order repaired and checked by reading, not compiled, not run on hardware
//
#include <hip/hip_runtime.h>
#include <hip/hip_bf16.h>

// ---------------- problem constants ----------------
constexpr int T  = 2048;   // tokens
constexpr int H  = 1024;   // hidden
constexpr int F  = 1024;   // ffn dim
constexpr int E  = 16;     // experts
constexpr int TM = 32;     // tokens per expert tile (2 WMMA M-tiles)
constexpr int XS = H + 8;  // padded LDS row stride (bf16 elems) -> 4-bank skew per row
constexpr int WAVES = 8;   // 256 threads / wave32

typedef __attribute__((ext_vector_type(16))) __bf16 v16bf;
typedef __attribute__((ext_vector_type(8)))  __bf16 v8bf;
typedef __attribute__((ext_vector_type(4)))  __bf16 v4bf;
typedef __attribute__((ext_vector_type(8)))  float  v8f;

__device__ __forceinline__ __bf16 f2bf(float f) {
  return static_cast<__bf16>(f);   // v_cvt_pk_bf16_f32 when paired (RNE)
}

// ---------------- init: zero output + per-expert counters ----------------
__global__ void k_init(float* __restrict__ out, int* __restrict__ counts) {
  int i = blockIdx.x * blockDim.x + threadIdx.x;
  if (i < T * H) out[i] = 0.f;
  if (i < E)     counts[i] = 0;
}

// ---------------- router: sigmoid scores, bias-corrected top-2, scatter ----------------
__global__ __launch_bounds__(256)
void k_router(const float* __restrict__ x, const float* __restrict__ gw,
              const float* __restrict__ bias,
              int* __restrict__ counts, int* __restrict__ tids, float* __restrict__ twts) {
  __shared__ float s_scr[16][16];
  __shared__ float s_cor[16][16];
  const int tid = threadIdx.x;
  const int e  = tid & 15;
  const int ty = tid >> 4;
  const int t  = blockIdx.x * 16 + ty;

  const float4* xr = (const float4*)(x + (size_t)t * H);
  const float4* wr = (const float4*)(gw + (size_t)e * H);
  float acc = 0.f;
  #pragma unroll 4
  for (int i = 0; i < H / 4; ++i) {
    float4 a = xr[i], b = wr[i];
    acc += a.x * b.x + a.y * b.y + a.z * b.z + a.w * b.w;
  }
  float score = 1.f / (1.f + __expf(-acc));
  s_scr[ty][e] = score;
  s_cor[ty][e] = score + bias[e];
  __syncthreads();

  if (e == 0) {
    int i0 = 0; float b0 = s_cor[ty][0];
    for (int j = 1; j < E; ++j) { float v = s_cor[ty][j]; if (v > b0) { b0 = v; i0 = j; } }
    int i1 = (i0 == 0) ? 1 : 0; float b1 = s_cor[ty][i1];
    for (int j = 0; j < E; ++j) {
      if (j == i0) continue;
      float v = s_cor[ty][j]; if (v > b1) { b1 = v; i1 = j; }
    }
    float w0 = s_scr[ty][i0], w1 = s_scr[ty][i1];
    float inv = 1.f / (w0 + w1 + 1e-20f);
    w0 *= inv; w1 *= inv;
    int s0 = atomicAdd(&counts[i0], 1);
    tids[i0 * T + s0] = t; twts[i0 * T + s0] = w0;
    int s1 = atomicAdd(&counts[i1], 1);
    tids[i1 * T + s1] = t; twts[i1 * T + s1] = w1;
  }
}

// ---------------- expert MLP: gate/up GEMM -> silu*up -> down GEMM -> scatter-add ----------------
__global__ __launch_bounds__(256, 1)
void k_moe(const float* __restrict__ x,
           const float* __restrict__ wg,   // [E][H][F]
           const float* __restrict__ wu,   // [E][H][F]
           const float* __restrict__ wd,   // [E][F][H]
           const int* __restrict__ counts,
           const int* __restrict__ tids,
           const float* __restrict__ twts,
           float* __restrict__ out) {
  const int e   = blockIdx.y;
  const int cnt = counts[e];
  const int m0  = blockIdx.x * TM;
  if (m0 >= cnt) return;
  const int nrows = min(TM, cnt - m0);

  extern __shared__ char smem[];
  __bf16* Xs = (__bf16*)smem;        // [TM][XS] gathered tokens (bf16)
  __bf16* Hs = Xs + TM * XS;         // [TM][XS] silu(g)*u*weight (bf16); also f32 gather staging
  __shared__ int   s_tid[TM];
  __shared__ float s_wt[TM];

  const int tid  = threadIdx.x;
  const int lane = tid & 31;
  const int wv   = tid >> 5;
  const int ln16 = lane & 15;
  const int lhi  = lane >> 4;        // 0: lanes 0-15, 1: lanes 16-31

  if (tid < TM) {
    if (tid < nrows) { s_tid[tid] = tids[e * T + m0 + tid]; s_wt[tid] = twts[e * T + m0 + tid]; }
    else             { s_tid[tid] = 0;                      s_wt[tid] = 0.f; }
  }
  __syncthreads();

  // ---- Gather token tile via async global->LDS DMA, then convert f32->bf16 ----
  // Stage 16 rows of f32 (64 KB) at a time into the (still unused) Hs region,
  // using GLOBAL_LOAD_ASYNC_TO_LDS_B128 (ASYNCcnt-tracked, no VGPR round trip),
  // then each thread converts exactly the slots it loaded into the bf16 Xs tile.
  {
    float* stage = (float*)Hs;                    // 16 rows x 1024 f32 = 64 KB <= |Hs|
    for (int c = 0; c < TM / 16; ++c) {
      for (int i = tid; i < 16 * 256; i += 256) { // 256 float4 segments per row
        const int row = i >> 8;
        const int seg = i & 255;
        const int gr  = c * 16 + row;
        if (gr < nrows) {
          unsigned long long gaddr =
              (unsigned long long)(uintptr_t)(x + (size_t)s_tid[gr] * H + seg * 4);
          unsigned ldsb = (unsigned)(size_t)(stage + row * 1024 + seg * 4);
          asm volatile("global_load_async_to_lds_b128 %0, %1, off"
                       :: "v"(ldsb), "v"(gaddr) : "memory");
        }
      }
      asm volatile("s_wait_asynccnt 0x0" ::: "memory");
      __syncthreads();
      for (int i = tid; i < 16 * 256; i += 256) {
        const int row = i >> 8;
        const int seg = i & 255;
        const int gr  = c * 16 + row;
        v4bf v;
        if (gr < nrows) {
          const float4 a = *(const float4*)(stage + row * 1024 + seg * 4);
          v[0] = f2bf(a.x); v[1] = f2bf(a.y); v[2] = f2bf(a.z); v[3] = f2bf(a.w);
        } else {
          v[0] = v[1] = v[2] = v[3] = f2bf(0.f);
        }
        *(v4bf*)&Xs[gr * XS + (seg << 2)] = v;
      }
      __syncthreads();                            // staging reused next chunk / by phase 1
    }
  }

  // A operand (16x32 bf16): lane<16 holds row M=lane, K 0..15;
  // lane>=16 holds row M=lane-16, K 16..31.  Two ds_load_b128.
  auto loadA = [&](const __bf16* base, int r0, int k0) -> v16bf {
    const __bf16* p = base + (r0 + ln16) * XS + k0 + lhi * 16;
    v8bf lo = *(const v8bf*)p;
    v8bf hi = *(const v8bf*)(p + 8);
    v16bf a;
    #pragma unroll
    for (int i = 0; i < 8; ++i) { a[i] = lo[i]; a[8 + i] = hi[i]; }
    return a;
  };

  // B operand (32x16 bf16): column N = lane&15; lanes 0-15 -> K 0..15,
  // lanes 16-31 -> K 16..31, K packed 2-per-VGPR.  Converted from f32 rows.
  auto loadB = [&](const float* w, int ldn, int n0, int k0) -> v16bf {
    const float* p = w + (size_t)(k0 + lhi * 16) * ldn + n0 + ln16;
    v16bf b;
    #pragma unroll
    for (int i = 0; i < 8; ++i) {
      b[2 * i]     = f2bf(p[(size_t)(2 * i)     * ldn]);
      b[2 * i + 1] = f2bf(p[(size_t)(2 * i + 1) * ldn]);
    }
    return b;
  };

  const float* wge = wg + (size_t)e * H * F;
  const float* wue = wu + (size_t)e * H * F;

  // ---- Phase 1: G = X*Wg, U = X*Wu, h = silu(G)*U * token_weight -> Hs ----
  for (int it = 0; it < F / (WAVES * 16); ++it) {
    const int n0 = (it * WAVES + wv) * 16;
    v8f g0 = {}, g1 = {}, u0 = {}, u1 = {};
    for (int k0 = 0; k0 < H; k0 += 32) {
      v16bf a0 = loadA(Xs, 0,  k0);
      v16bf a1 = loadA(Xs, 16, k0);
      v16bf bg = loadB(wge, F, n0, k0);
      v16bf bu = loadB(wue, F, n0, k0);
      if (k0 + 32 < H) {
        __builtin_prefetch(wge + (size_t)(k0 + 32 + lhi * 16) * F + n0 + ln16, 0, 0);
        __builtin_prefetch(wue + (size_t)(k0 + 32 + lhi * 16) * F + n0 + ln16, 0, 0);
      }
      g0 = __builtin_amdgcn_wmma_f32_16x16x32_bf16(false, a0, false, bg, (short)0, g0, false, false);
      g1 = __builtin_amdgcn_wmma_f32_16x16x32_bf16(false, a1, false, bg, (short)0, g1, false, false);
      u0 = __builtin_amdgcn_wmma_f32_16x16x32_bf16(false, a0, false, bu, (short)0, u0, false, false);
      u1 = __builtin_amdgcn_wmma_f32_16x16x32_bf16(false, a1, false, bu, (short)0, u1, false, false);
    }
    // C/D layout: element (M,N): lane = N + 16*(M>=8), vgpr = M&7.
    #pragma unroll
    for (int r = 0; r < 8; ++r) {
      const int mA = r + lhi * 8;
      float h0 = g0[r] / (1.f + __expf(-g0[r])) * u0[r] * s_wt[mA];
      float h1 = g1[r] / (1.f + __expf(-g1[r])) * u1[r] * s_wt[16 + mA];
      Hs[mA * XS + n0 + ln16]        = f2bf(h0);
      Hs[(16 + mA) * XS + n0 + ln16] = f2bf(h1);
    }
  }
  __syncthreads();

  // ---- Phase 2: Y = h * Wd, scatter-accumulate into out (weights already folded) ----
  const float* wde = wd + (size_t)e * F * H;
  for (int it = 0; it < H / (WAVES * 16); ++it) {
    const int n0 = (it * WAVES + wv) * 16;
    v8f y0 = {}, y1 = {};
    for (int k0 = 0; k0 < F; k0 += 32) {
      v16bf a0 = loadA(Hs, 0,  k0);
      v16bf a1 = loadA(Hs, 16, k0);
      v16bf bd = loadB(wde, H, n0, k0);
      if (k0 + 32 < F)
        __builtin_prefetch(wde + (size_t)(k0 + 32 + lhi * 16) * H + n0 + ln16, 0, 0);
      y0 = __builtin_amdgcn_wmma_f32_16x16x32_bf16(false, a0, false, bd, (short)0, y0, false, false);
      y1 = __builtin_amdgcn_wmma_f32_16x16x32_bf16(false, a1, false, bd, (short)0, y1, false, false);
    }
    #pragma unroll
    for (int r = 0; r < 8; ++r) {
      const int mA = r + lhi * 8;
      if (mA < nrows)
        atomicAdd(&out[(size_t)s_tid[mA] * H + n0 + ln16], y0[r]);
      if (16 + mA < nrows)
        atomicAdd(&out[(size_t)s_tid[16 + mA] * H + n0 + ln16], y1[r]);
    }
  }
}

// ---------------- host-side launch ----------------
extern "C" void kernel_launch(void* const* d_in, const int* in_sizes, int n_in,
                              void* d_out, int out_size, void* d_ws, size_t ws_size,
                              hipStream_t stream) {
  const float* x    = (const float*)d_in[0];
  const float* gw   = (const float*)d_in[1];
  const float* wg   = (const float*)d_in[2];
  const float* wu   = (const float*)d_in[3];
  const float* wd   = (const float*)d_in[4];
  const float* bias = (const float*)d_in[5];
  float* out = (float*)d_out;

  char* ws = (char*)d_ws;
  int*   counts = (int*)ws;                          // E counters
  int*   tids   = (int*)(ws + 256);                  // E*T token ids
  float* twts   = (float*)(ws + 256 + (size_t)E * T * sizeof(int)); // E*T weights

  k_init<<<(T * H + 255) / 256, 256, 0, stream>>>(out, counts);
  k_router<<<T / 16, 256, 0, stream>>>(x, gw, bias, counts, tids, twts);

  dim3 grid(T / TM, E);
  size_t lds = (size_t)2 * TM * XS * sizeof(__bf16); // Xs + Hs = 132 KB (<= 320 KB/WGP)
  k_moe<<<grid, 256, lds, stream>>>(x, wg, wu, wd, counts, tids, twts, out);
}
